// LocalEncoder_31799937860250
// MI455X (gfx1250) — matrix operand order repaired and checked by reading, not compiled, not run on hardware
//
#include <hip/hip_runtime.h>
#include <hip/hip_bf16.h>
#include <stdint.h>

// ---------------- problem constants (match reference) ----------------
#define NN 50000
#define EE 800000
#define DD 64

// ---------------- input index map (setup_inputs dict order, leaves DFS) ----
enum {
  I_X = 0, I_EA = 1, I_ROT = 2,
  I_C_L1W = 3, I_C_L1B, I_C_N1G, I_C_N1B, I_C_L2W, I_C_L2B, I_C_N2G, I_C_N2B,
  I_C_L3W, I_C_L3B, I_C_N3G, I_C_N3B,                       // 3..14
  I_A_L1W = 15, I_A_L1B, I_A_N1G, I_A_N1B, I_A_L2W, I_A_L2B, // 15..20
  I_B_L1W = 21, I_B_L1B, I_B_N1G, I_B_N1B, I_B_L2W, I_B_L2B, // 21..26
  I_G_N1G = 27, I_G_N1B, I_G_LW, I_G_LB, I_G_N2G, I_G_N2B,   // 27..32
  I_QW = 33, I_QB, I_KW, I_KB, I_VW, I_VB,                   // 33..38
  I_SW_ = 39, I_SB_, I_IHW, I_IHB, I_HHW, I_HHB, I_OW, I_OB, // 39..46
  I_N1G = 47, I_N1B, I_N2G, I_N2B,                           // 47..50
  I_M1W = 51, I_M1B, I_M2W, I_M2B,                           // 51..54
  I_BOS = 55, I_EI = 56, I_BOSM = 57, I_T = 58
};

struct InPtrs { const void* p[64]; };

#define PF(i)  ((const float*)ip.p[(i)])
#define PI(i)  ((const int*)ip.p[(i)])
#define PU8(i) ((const unsigned char*)ip.p[(i)])

typedef _Float16 v16h __attribute__((ext_vector_type(16)));
typedef _Float16 v8h  __attribute__((ext_vector_type(8)));
typedef float    v8f  __attribute__((ext_vector_type(8)));

// ================= WMMA fragment helpers (wave32, 16x16x32 f16) =========
// A (16x32 f16): lane<16 holds row M=lane, elems 0..7 -> K 0..7, 8..15 -> K 16..23
//                lane>=16 holds row M=lane-16, elems 0..7 -> K 8..15, 8..15 -> K 24..31
__device__ __forceinline__ v16h load_a_frag(const _Float16* act, int SA, int k0, int lane) {
  int row = lane & 15, hi = lane >> 4;
  const _Float16* p0 = act + row * SA + k0 + hi * 8;
  v8h a0 = *(const v8h*)p0;
  v8h a1 = *(const v8h*)(p0 + 16);
  v16h a;
#pragma unroll
  for (int i = 0; i < 8; ++i) { a[i] = a0[i]; a[8 + i] = a1[i]; }
  return a;
}

// B (32x16 f16, K x N), weights stored transposed wT[n][k] (k contiguous):
// lane<16: N=lane, elems 0..7 -> K 0..7, 8..15 -> K 16..23; lane>=16: N=lane-16, elems 0..7 -> K 8..15, 8..15 -> K 24..31
__device__ __forceinline__ v16h load_b_frag(const _Float16* wT, int SW, int n0, int k0, int lane) {
  int n = lane & 15, kh = lane >> 4;
  const _Float16* p = wT + (size_t)(n0 + n) * SW + k0 + kh * 8;
  v8h b0 = *(const v8h*)p;
  v8h b1 = *(const v8h*)(p + 16);
  v16h b;
#pragma unroll
  for (int i = 0; i < 8; ++i) { b[i] = b0[i]; b[8 + i] = b1[i]; }
  return b;
}

__device__ __forceinline__ v8f bias_acc(const float* bias, int n0, int lane) {
  int n = lane & 15;
  float bv = bias ? bias[n0 + n] : 0.f;
  v8f a;
#pragma unroll
  for (int r = 0; r < 8; ++r) a[r] = bv;
  return a;
}

__device__ __forceinline__ v8f gemm_acc(const _Float16* act, int SA, const _Float16* wT, int SW,
                                        int n0, int KT, v8f acc, int lane) {
  for (int kt = 0; kt < KT; ++kt) {
    v16h a = load_a_frag(act, SA, kt * 32, lane);
    v16h b = load_b_frag(wT, SW, n0, kt * 32, lane);
    acc = __builtin_amdgcn_wmma_f32_16x16x32_f16(false, a, false, b, (short)0, acc, false, false);
  }
  return acc;
}

// full 16xK @ KxN GEMM tile -> f32 LDS (optionally accumulate into out)
__device__ __forceinline__ void gemm_f32(const _Float16* act, int SA, const _Float16* wT, int SW,
                                         const float* bias, int NT, int KT, float* out, int SO,
                                         bool accumulate, int lane) {
  int n = lane & 15, mb = (lane >> 4) * 8;
  for (int nt = 0; nt < NT; ++nt) {
    v8f acc = bias_acc(bias, nt * 16, lane);
    acc = gemm_acc(act, SA, wT, SW, nt * 16, KT, acc, lane);
#pragma unroll
    for (int r = 0; r < 8; ++r) {
      float v = acc[r];
      if (accumulate) v += out[(mb + r) * SO + nt * 16 + n];
      out[(mb + r) * SO + nt * 16 + n] = v;
    }
  }
}

// GEMM tile -> fp16 LDS with ReLU epilogue (for FFN hidden)
__device__ __forceinline__ void gemm_h16relu(const _Float16* act, int SA, const _Float16* wT, int SW,
                                             const float* bias, int NT, int KT, _Float16* out, int SO,
                                             int lane) {
  int n = lane & 15, mb = (lane >> 4) * 8;
  for (int nt = 0; nt < NT; ++nt) {
    v8f acc = bias_acc(bias, nt * 16, lane);
    acc = gemm_acc(act, SA, wT, SW, nt * 16, KT, acc, lane);
#pragma unroll
    for (int r = 0; r < 8; ++r)
      out[(mb + r) * SO + nt * 16 + n] = (_Float16)fmaxf(acc[r], 0.f);
  }
}

// LayerNorm over 64 dims for a 16-row f32 LDS tile; 2 lanes per row, shfl combine
__device__ __forceinline__ void ln16(float* buf, int S, const float* g, const float* beta, int lane) {
  int r = lane & 15, h = lane >> 4;
  float* row = buf + r * S + h * 32;
  float s = 0.f, s2 = 0.f;
#pragma unroll
  for (int i = 0; i < 32; ++i) { float x = row[i]; s += x; s2 += x * x; }
  s += __shfl_xor(s, 16, 32);
  s2 += __shfl_xor(s2, 16, 32);
  float m = s * 0.015625f;
  float var = s2 * 0.015625f - m * m;
  float rs = __frsqrt_rn(var + 1e-5f);
#pragma unroll
  for (int i = 0; i < 32; ++i) { int c = h * 32 + i; row[i] = (row[i] - m) * rs * g[c] + beta[c]; }
}

// f32 tile -> fp16 tile (optional ReLU), 16x64 elems over 32 lanes
__device__ __forceinline__ void to_h16(const float* in, int SI, _Float16* out, int SO, bool relu, int lane) {
#pragma unroll
  for (int i = 0; i < 32; ++i) {
    int idx = lane + (i << 5);
    int r = idx >> 6, c = idx & 63;
    float v = in[r * SI + c];
    if (relu) v = fmaxf(v, 0.f);
    out[r * SO + c] = (_Float16)v;
  }
}

// cooperative loaders
__device__ __forceinline__ void load_f32(const float* src, int n, float* dst, int tid, int nth) {
  for (int i = tid; i < n; i += nth) dst[i] = src[i];
}
// w row-major (K x N) -> wT[n][k] fp16 with row stride SW
__device__ __forceinline__ void load_wT(const float* w, int K, int Nc, _Float16* wT, int SW, int tid, int nth) {
  for (int idx = tid; idx < K * Nc; idx += nth) {
    int i = idx / Nc, o = idx % Nc;
    wT[(size_t)o * SW + i] = (_Float16)w[idx];
  }
}

// =========================== K0: init scatter state ======================
__global__ void k0_init(float* amax, float* asum, float* agg) {
  int idx = blockIdx.x * blockDim.x + threadIdx.x;
  if (idx < NN * 8) amax[idx] = -3.0e38f;
  else if (idx < NN * 16) asum[idx - NN * 8] = 0.f;
  else if (idx < NN * 80) agg[idx - NN * 16] = 0.f;
}

// =========================== K1: center embed + norm1 ====================
// block 128 = 4 waves, each wave one 16-node tile
__global__ void k1_center(InPtrs ip, float* ce_ws, float* cn_ws) {
  extern __shared__ char smem[];
  float* fsm = (float*)smem;
  float* W1 = fsm;           // 128
  float* b1 = W1 + 128;      // 64 each below
  float* g1 = b1 + 64;  float* e1 = g1 + 64;
  float* b2 = e1 + 64;  float* g2 = b2 + 64;  float* e2 = g2 + 64;
  float* b3 = e2 + 64;  float* g3 = b3 + 64;  float* e3 = g3 + 64;
  float* gn = e3 + 64;  float* en = gn + 64;
  float* bos = en + 64;      // 64 -> 896 floats total
  _Float16* wT2 = (_Float16*)(fsm + 896);
  _Float16* wT3 = wT2 + 64 * 72;
  char* pw = (char*)(wT3 + 64 * 72);
  int tid = threadIdx.x, lane = tid & 31, wv = tid >> 5;
  char* wvb = pw + wv * 6912;
  _Float16* act = (_Float16*)wvb;
  float* fbuf = (float*)(wvb + 2304);

  load_f32(PF(I_C_L1W), 128, W1, tid, 128);
  load_f32(PF(I_C_L1B), 64, b1, tid, 128);
  load_f32(PF(I_C_N1G), 64, g1, tid, 128);  load_f32(PF(I_C_N1B), 64, e1, tid, 128);
  load_f32(PF(I_C_L2B), 64, b2, tid, 128);
  load_f32(PF(I_C_N2G), 64, g2, tid, 128);  load_f32(PF(I_C_N2B), 64, e2, tid, 128);
  load_f32(PF(I_C_L3B), 64, b3, tid, 128);
  load_f32(PF(I_C_N3G), 64, g3, tid, 128);  load_f32(PF(I_C_N3B), 64, e3, tid, 128);
  load_f32(PF(I_N1G), 64, gn, tid, 128);    load_f32(PF(I_N1B), 64, en, tid, 128);
  { int t = *PI(I_T); load_f32(PF(I_BOS) + (size_t)t * 64, 64, bos, tid, 128); }
  load_wT(PF(I_C_L2W), 64, 64, wT2, 72, tid, 128);
  load_wT(PF(I_C_L3W), 64, 64, wT3, 72, tid, 128);
  __syncthreads();

  int n0 = blockIdx.x * 64 + wv * 16;
  // layer1: rotate + 2->64 linear
#pragma unroll
  for (int i = 0; i < 32; ++i) {
    int idx = lane + (i << 5); int r = idx >> 6, c = idx & 63;
    int nd = n0 + r; int nc = nd < NN ? nd : NN - 1;
    const float* R = PF(I_ROT) + (size_t)nc * 4;
    float x0 = PF(I_X)[nc * 2], x1 = PF(I_X)[nc * 2 + 1];
    float xr0 = x0 * R[0] + x1 * R[2];
    float xr1 = x0 * R[1] + x1 * R[3];
    fbuf[r * 72 + c] = xr0 * W1[c] + xr1 * W1[64 + c] + b1[c];
  }
  __syncthreads();
  ln16(fbuf, 72, g1, e1, lane); __syncthreads();
  to_h16(fbuf, 72, act, 72, true, lane); __syncthreads();
  gemm_f32(act, 72, wT2, 72, b2, 4, 2, fbuf, 72, false, lane); __syncthreads();
  ln16(fbuf, 72, g2, e2, lane); __syncthreads();
  to_h16(fbuf, 72, act, 72, true, lane); __syncthreads();
  gemm_f32(act, 72, wT3, 72, b3, 4, 2, fbuf, 72, false, lane); __syncthreads();
  ln16(fbuf, 72, g3, e3, lane); __syncthreads();
  // bos substitution + store ce
#pragma unroll
  for (int i = 0; i < 32; ++i) {
    int idx = lane + (i << 5); int r = idx >> 6, c = idx & 63;
    int nd = n0 + r; int nc = nd < NN ? nd : NN - 1;
    float v = fbuf[r * 72 + c];
    if (PU8(I_BOSM)[nc]) v = bos[c];
    fbuf[r * 72 + c] = v;
    if (nd < NN) ce_ws[(size_t)nd * 64 + c] = v;
  }
  __syncthreads();
  ln16(fbuf, 72, gn, en, lane); __syncthreads();
#pragma unroll
  for (int i = 0; i < 32; ++i) {
    int idx = lane + (i << 5); int r = idx >> 6, c = idx & 63;
    int nd = n0 + r;
    if (nd < NN) cn_ws[(size_t)nd * 64 + c] = fbuf[r * 72 + c];
  }
}

// ================== K2: fused edge message / q,k,v / alpha ==============
// block 128 = 4 waves, each wave one 16-edge tile (E % 64 == 0)
__global__ void k2_edges(InPtrs ip, const float* cn_ws, _Float16* vws, float* alpha_ws, float* amax) {
  extern __shared__ char smem[];
  float* fsm = (float*)smem;
  float* W1a = fsm;          // 128
  float* b1a = W1a + 128; float* g1a = b1a + 64; float* e1a = g1a + 64;
  float* W1b = e1a + 64;     // 128
  float* b1b = W1b + 128; float* g1b = b1b + 64; float* e1b = g1b + 64;
  float* gs1 = e1b + 64;  float* es1 = gs1 + 64;
  float* bAB = es1 + 64;     // combined l2a.b + l2b.b
  float* bL  = bAB + 64;  float* gs2 = bL + 64; float* es2 = gs2 + 64;
  float* bq  = es2 + 64;  float* bk  = bq + 64; float* bv  = bk + 64;  // -> 1216 floats
  _Float16* wTa2 = (_Float16*)(fsm + 1216);
  _Float16* wTb2 = wTa2 + 4608;
  _Float16* wTl  = wTb2 + 4608;
  _Float16* wTq  = wTl  + 4608;
  _Float16* wTk  = wTq  + 4608;
  _Float16* wTv  = wTk  + 4608;
  char* pw = (char*)(wTv + 4608);
  int tid = threadIdx.x, lane = tid & 31, wv = tid >> 5;
  char* wvb = pw + wv * 11840;
  float* xj = (float*)wvb;     // 32
  float* er = xj + 32;         // 32
  int* dst16 = (int*)(er + 32);
  _Float16* act = (_Float16*)(wvb + 320);
  float* fA = (float*)(wvb + 2624);
  float* fB = (float*)(wvb + 7232);

  load_f32(PF(I_A_L1W), 128, W1a, tid, 128);
  load_f32(PF(I_A_L1B), 64, b1a, tid, 128);
  load_f32(PF(I_A_N1G), 64, g1a, tid, 128); load_f32(PF(I_A_N1B), 64, e1a, tid, 128);
  load_f32(PF(I_B_L1W), 128, W1b, tid, 128);
  load_f32(PF(I_B_L1B), 64, b1b, tid, 128);
  load_f32(PF(I_B_N1G), 64, g1b, tid, 128); load_f32(PF(I_B_N1B), 64, e1b, tid, 128);
  load_f32(PF(I_G_N1G), 64, gs1, tid, 128); load_f32(PF(I_G_N1B), 64, es1, tid, 128);
  for (int i = tid; i < 64; i += 128) bAB[i] = PF(I_A_L2B)[i] + PF(I_B_L2B)[i];
  load_f32(PF(I_G_LB), 64, bL, tid, 128);
  load_f32(PF(I_G_N2G), 64, gs2, tid, 128); load_f32(PF(I_G_N2B), 64, es2, tid, 128);
  load_f32(PF(I_QB), 64, bq, tid, 128);
  load_f32(PF(I_KB), 64, bk, tid, 128);
  load_f32(PF(I_VB), 64, bv, tid, 128);
  load_wT(PF(I_A_L2W), 64, 64, wTa2, 72, tid, 128);
  load_wT(PF(I_B_L2W), 64, 64, wTb2, 72, tid, 128);
  load_wT(PF(I_G_LW),  64, 64, wTl,  72, tid, 128);
  load_wT(PF(I_QW),    64, 64, wTq,  72, tid, 128);
  load_wT(PF(I_KW),    64, 64, wTk,  72, tid, 128);
  load_wT(PF(I_VW),    64, 64, wTv,  72, tid, 128);
  __syncthreads();

  int e0 = blockIdx.x * 64 + wv * 16;
  if (lane < 16) {
    int e = e0 + lane;
    int s = PI(I_EI)[e], d = PI(I_EI)[EE + e];
    const float* R = PF(I_ROT) + (size_t)d * 4;
    float x0 = PF(I_X)[s * 2], x1 = PF(I_X)[s * 2 + 1];
    float a0 = PF(I_EA)[(size_t)e * 2], a1 = PF(I_EA)[(size_t)e * 2 + 1];
    xj[lane * 2] = x0 * R[0] + x1 * R[2];  xj[lane * 2 + 1] = x0 * R[1] + x1 * R[3];
    er[lane * 2] = a0 * R[0] + a1 * R[2];  er[lane * 2 + 1] = a0 * R[1] + a1 * R[3];
    dst16[lane] = d;
  }
  __syncthreads();
  // branch a: 2->64 lin, LN, ReLU, 64->64 WMMA -> fB (bias = b2a+b2b)
#pragma unroll
  for (int i = 0; i < 32; ++i) {
    int idx = lane + (i << 5); int r = idx >> 6, c = idx & 63;
    fA[r * 72 + c] = xj[r * 2] * W1a[c] + xj[r * 2 + 1] * W1a[64 + c] + b1a[c];
  }
  __syncthreads();
  ln16(fA, 72, g1a, e1a, lane); __syncthreads();
  to_h16(fA, 72, act, 72, true, lane); __syncthreads();
  gemm_f32(act, 72, wTa2, 72, bAB, 4, 2, fB, 72, false, lane); __syncthreads();
  // branch b accumulated into fB
#pragma unroll
  for (int i = 0; i < 32; ++i) {
    int idx = lane + (i << 5); int r = idx >> 6, c = idx & 63;
    fA[r * 72 + c] = er[r * 2] * W1b[c] + er[r * 2 + 1] * W1b[64 + c] + b1b[c];
  }
  __syncthreads();
  ln16(fA, 72, g1b, e1b, lane); __syncthreads();
  to_h16(fA, 72, act, 72, true, lane); __syncthreads();
  gemm_f32(act, 72, wTb2, 72, nullptr, 4, 2, fB, 72, true, lane); __syncthreads();
  // aggr: LN, ReLU, linear, LN -> nbr (fp16 in act)
  ln16(fB, 72, gs1, es1, lane); __syncthreads();
  to_h16(fB, 72, act, 72, true, lane); __syncthreads();
  gemm_f32(act, 72, wTl, 72, bL, 4, 2, fA, 72, false, lane); __syncthreads();
  ln16(fA, 72, gs2, es2, lane); __syncthreads();
  to_h16(fA, 72, act, 72, false, lane); __syncthreads();
  // v: WMMA -> fp16 straight to global workspace
  {
    int n = lane & 15, mb = (lane >> 4) * 8;
    for (int nt = 0; nt < 4; ++nt) {
      v8f acc = bias_acc(bv, nt * 16, lane);
      acc = gemm_acc(act, 72, wTv, 72, nt * 16, 2, acc, lane);
#pragma unroll
      for (int r = 0; r < 8; ++r)
        vws[(size_t)(e0 + mb + r) * 64 + nt * 16 + n] = (_Float16)acc[r];
    }
  }
  // k -> fA
  gemm_f32(act, 72, wTk, 72, bk, 4, 2, fA, 72, false, lane); __syncthreads();
  // q from cn[dst] -> fB
#pragma unroll
  for (int i = 0; i < 32; ++i) {
    int idx = lane + (i << 5); int r = idx >> 6, c = idx & 63;
    act[r * 72 + c] = (_Float16)cn_ws[(size_t)dst16[r] * 64 + c];
  }
  __syncthreads();
  gemm_f32(act, 72, wTq, 72, bq, 4, 2, fB, 72, false, lane); __syncthreads();
  // alpha[e][h] = (q.k)/sqrt(8) ; scatter-max into amax
#pragma unroll
  for (int i = 0; i < 4; ++i) {
    int p = lane + (i << 5);      // 0..127 = 16 edges x 8 heads
    int el = p >> 3, h = p & 7;
    float dot = 0.f;
#pragma unroll
    for (int d = 0; d < 8; ++d)
      dot += fB[el * 72 + h * 8 + d] * fA[el * 72 + h * 8 + d];
    float al = dot * 0.3535533905932738f;
    int e = e0 + el;
    alpha_ws[(size_t)e * 8 + h] = al;
    __hip_atomic_fetch_max(&amax[(size_t)dst16[el] * 8 + h], al,
                           __ATOMIC_RELAXED, __HIP_MEMORY_SCOPE_AGENT);
  }
}

// ================= K3: exp(alpha - max) + segment sum ====================
__global__ void k3_expsum(const int* ei, float* alpha_ws, const float* amax, float* asum) {
  int idx = blockIdx.x * blockDim.x + threadIdx.x;
  if (idx >= EE * 8) return;
  int e = idx >> 3, h = idx & 7;
  int d = ei[EE + e];
  float a = __expf(alpha_ws[idx] - amax[(size_t)d * 8 + h]);
  alpha_ws[idx] = a;
  atomicAdd(&asum[(size_t)d * 8 + h], a);
}

// ================= K4: weighted scatter of v into agg ====================
__global__ void k4_scatter(const int* ei, const float* alpha_ws, const float* asum,
                           const _Float16* vws, float* agg) {
  int e = blockIdx.x * blockDim.x + threadIdx.x;
  if (e >= EE) return;
  int d = ei[EE + e];
  float w[8];
#pragma unroll
  for (int h = 0; h < 8; ++h)
    w[h] = alpha_ws[(size_t)e * 8 + h] / (asum[(size_t)d * 8 + h] + 1e-16f);
  const _Float16* vr = vws + (size_t)e * 64;
  float* ar = agg + (size_t)d * 64;
#pragma unroll
  for (int c = 0; c < 64; ++c)
    atomicAdd(&ar[c], w[c >> 3] * (float)vr[c]);
}

// ================= K5: gated update + FFN ================================
__global__ void k5_update(InPtrs ip, const float* ce_ws, const float* cn_ws,
                          const float* agg, float* out) {
  extern __shared__ char smem[];
  float* fsm = (float*)smem;
  float* b_ihh = fsm;               // combined ih.b + hh.b
  float* b_self = b_ihh + 64;
  float* b_out  = b_self + 64;
  float* gn2 = b_out + 64; float* en2 = gn2 + 64;
  float* bm1 = en2 + 64;            // 256
  float* bm2 = bm1 + 256;           // 64 -> 640 floats
  _Float16* wTih = (_Float16*)(fsm + 640);
  _Float16* wThh = wTih + 4608;
  _Float16* wTsf = wThh + 4608;
  _Float16* wTot = wTsf + 4608;
  _Float16* wTm1 = wTot + 4608;     // 256 x 72
  _Float16* wTm2 = wTm1 + 18432;    // 64 x 264
  char* pw = (char*)(wTm2 + 16896);
  int tid = threadIdx.x, lane = tid & 31, wv = tid >> 5;
  char* wvb = pw + wv * 19968;
  _Float16* act  = (_Float16*)wvb;          // 16x72
  _Float16* a256 = (_Float16*)(wvb + 2304); // 16x264
  float* fA = (float*)(wvb + 10752);
  float* fB = (float*)(wvb + 15360);

  for (int i = tid; i < 64; i += 128) b_ihh[i] = PF(I_IHB)[i] + PF(I_HHB)[i];
  load_f32(PF(I_SB_), 64, b_self, tid, 128);
  load_f32(PF(I_OB), 64, b_out, tid, 128);
  load_f32(PF(I_N2G), 64, gn2, tid, 128); load_f32(PF(I_N2B), 64, en2, tid, 128);
  load_f32(PF(I_M1B), 256, bm1, tid, 128);
  load_f32(PF(I_M2B), 64, bm2, tid, 128);
  load_wT(PF(I_IHW), 64, 64, wTih, 72, tid, 128);
  load_wT(PF(I_HHW), 64, 64, wThh, 72, tid, 128);
  load_wT(PF(I_SW_), 64, 64, wTsf, 72, tid, 128);
  load_wT(PF(I_OW),  64, 64, wTot, 72, tid, 128);
  load_wT(PF(I_M1W), 64, 256, wTm1, 72, tid, 128);
  load_wT(PF(I_M2W), 256, 64, wTm2, 264, tid, 128);
  __syncthreads();

  int n0 = blockIdx.x * 64 + wv * 16;
  int n = lane & 15, mb = (lane >> 4) * 8;
  // agg -> fA (f32) + act (fp16)
#pragma unroll
  for (int i = 0; i < 32; ++i) {
    int idx = lane + (i << 5); int r = idx >> 6, c = idx & 63;
    int nd = n0 + r; int nc = nd < NN ? nd : NN - 1;
    float v = agg[(size_t)nc * 64 + c];
    fA[r * 72 + c] = v;
    act[r * 72 + c] = (_Float16)v;
  }
  __syncthreads();
  v8f aG[4], aS[4];
  for (int nt = 0; nt < 4; ++nt) {
    aG[nt] = bias_acc(b_ihh, nt * 16, lane);
    aG[nt] = gemm_acc(act, 72, wTih, 72, nt * 16, 2, aG[nt], lane);
  }
  __syncthreads();
  // act = cn
#pragma unroll
  for (int i = 0; i < 32; ++i) {
    int idx = lane + (i << 5); int r = idx >> 6, c = idx & 63;
    int nd = n0 + r; int nc = nd < NN ? nd : NN - 1;
    act[r * 72 + c] = (_Float16)cn_ws[(size_t)nc * 64 + c];
  }
  __syncthreads();
  for (int nt = 0; nt < 4; ++nt) {
    aG[nt] = gemm_acc(act, 72, wThh, 72, nt * 16, 2, aG[nt], lane);
    aS[nt] = bias_acc(b_self, nt * 16, lane);
    aS[nt] = gemm_acc(act, 72, wTsf, 72, nt * 16, 2, aS[nt], lane);
  }
  // upd = agg + sigmoid(gate)*(self - agg) -> fB
  for (int nt = 0; nt < 4; ++nt) {
#pragma unroll
    for (int r = 0; r < 8; ++r) {
      float gp = aG[nt][r], sv = aS[nt][r];
      float ag = fA[(mb + r) * 72 + nt * 16 + n];
      float gate = 1.f / (1.f + __expf(-gp));
      fB[(mb + r) * 72 + nt * 16 + n] = ag + gate * (sv - ag);
    }
  }
  __syncthreads();
  to_h16(fB, 72, act, 72, false, lane); __syncthreads();
  // ce = ce_old + out_proj(upd) -> fA
  for (int nt = 0; nt < 4; ++nt) {
    v8f a = bias_acc(b_out, nt * 16, lane);
    a = gemm_acc(act, 72, wTot, 72, nt * 16, 2, a, lane);
#pragma unroll
    for (int r = 0; r < 8; ++r) {
      int m = mb + r; int nd = n0 + m; int nc = nd < NN ? nd : NN - 1;
      fA[m * 72 + nt * 16 + n] = ce_ws[(size_t)nc * 64 + nt * 16 + n] + a[r];
    }
  }
  __syncthreads();
  // FFN: h = LN(ce); ce += mlp2(relu(mlp1(h)))
#pragma unroll
  for (int i = 0; i < 32; ++i) {
    int idx = lane + (i << 5); int r = idx >> 6, c = idx & 63;
    fB[r * 72 + c] = fA[r * 72 + c];
  }
  __syncthreads();
  ln16(fB, 72, gn2, en2, lane); __syncthreads();
  to_h16(fB, 72, act, 72, false, lane); __syncthreads();
  gemm_h16relu(act, 72, wTm1, 72, bm1, 16, 2, a256, 264, lane); __syncthreads();
  gemm_f32(a256, 264, wTm2, 264, bm2, 4, 8, fB, 72, false, lane); __syncthreads();
#pragma unroll
  for (int i = 0; i < 32; ++i) {
    int idx = lane + (i << 5); int r = idx >> 6, c = idx & 63;
    int nd = n0 + r;
    if (nd < NN) out[(size_t)nd * 64 + c] = fA[r * 72 + c] + fB[r * 72 + c];
  }
}

// =========================== host launcher ===============================
extern "C" void kernel_launch(void* const* d_in, const int* in_sizes, int n_in,
                              void* d_out, int out_size, void* d_ws, size_t ws_size,
                              hipStream_t stream) {
  (void)in_sizes; (void)out_size; (void)ws_size;
  InPtrs ip{};
  for (int i = 0; i < n_in && i < 64; ++i) ip.p[i] = d_in[i];

  // workspace carve (~170 MB):
  float* ce = (float*)d_ws;                           // N*64 f32
  float* cn = ce + (size_t)NN * 64;                   // N*64 f32
  _Float16* vws = (_Float16*)(cn + (size_t)NN * 64);  // E*64 fp16
  float* alpha = (float*)(vws + (size_t)EE * 64);     // E*8 f32
  float* amax = alpha + (size_t)EE * 8;               // N*8 f32
  float* asum = amax + (size_t)NN * 8;                // N*8 f32
  float* agg  = asum + (size_t)NN * 8;                // N*64 f32

  const size_t K1_SH = 49664;
  const size_t K2_SH = 107520;
  const size_t K5_SH = 189952;

  k0_init<<<(NN * 80 + 255) / 256, 256, 0, stream>>>(amax, asum, agg);
  k1_center<<<(NN + 63) / 64, 128, K1_SH, stream>>>(ip, ce, cn);
  k2_edges<<<EE / 64, 128, K2_SH, stream>>>(ip, cn, vws, alpha, amax);
  k3_expsum<<<(EE * 8 + 255) / 256, 256, 0, stream>>>((const int*)d_in[I_EI], alpha, amax, asum);
  k4_scatter<<<(EE + 255) / 256, 256, 0, stream>>>((const int*)d_in[I_EI], alpha, asum, vws, agg);
  k5_update<<<(NN + 63) / 64, 128, K5_SH, stream>>>(ip, ce, cn, agg, (float*)d_out);
}
